// EdgeClassifier_41815801594236
// MI455X (gfx1250) — compile-verified, matrix operand order checked by reading, not run on hardware
//
#include <hip/hip_runtime.h>
#include <hip/hip_fp16.h>
#include <cmath>

// ---------------------------------------------------------------------------
// CDNA5 (gfx1250) GNN edge classifier.
//   - wave32 WMMA f32_16x16x32_f16 for the two big GEMMs
//   - f16 node embeddings gathered directly into A fragments (no edge_emb)
//   - global_atomic_add_f32 scatter for GCN aggregation
// ---------------------------------------------------------------------------

typedef __attribute__((ext_vector_type(16))) _Float16 v16h;
typedef __attribute__((ext_vector_type(8)))  _Float16 v8h;
typedef __attribute__((ext_vector_type(8)))  float    v8f;

#define N_NODES 50000
#define N_EDGES 800000
#define HID     128
#define NODE_F  5
#define EDGE_F  4
#define KCAT    (2 * HID + EDGE_F)   // 260
#define NKT_MLP 9                    // K padded to 288 = 9*32
#define NKT_L2  4                    // K = 128 = 4*32
#define NNT     8                    // 128 cols / 16

// ---------------- elementwise / setup kernels ------------------------------

__global__ void k_zero(float* __restrict__ p, int n) {
  int i = blockIdx.x * blockDim.x + threadIdx.x;
  if (i < n) p[i] = 0.f;
}

__global__ void k_deg(const int* __restrict__ dst, float* __restrict__ deg) {
  int e = blockIdx.x * blockDim.x + threadIdx.x;
  if (e < N_EDGES) unsafeAtomicAdd(&deg[dst[e]], 1.f);
}

// deg -> dis = rsqrt(deg + 1)   (self-loop adds 1, so always > 0)
__global__ void k_dis(float* __restrict__ deg) {
  int n = blockIdx.x * blockDim.x + threadIdx.x;
  if (n < N_NODES) deg[n] = rsqrtf(deg[n] + 1.f);
}

__global__ void k_norm(const int* __restrict__ src, const int* __restrict__ dst,
                       const float* __restrict__ dis, float* __restrict__ nrm) {
  int e = blockIdx.x * blockDim.x + threadIdx.x;
  if (e < N_EDGES) nrm[e] = dis[src[e]] * dis[dst[e]];
}

// xw = x @ W1   (K = 5, plain VALU)
__global__ void k_xw1(const float* __restrict__ x, const float* __restrict__ W1,
                      float* __restrict__ xw) {
  int idx = blockIdx.x * blockDim.x + threadIdx.x;
  if (idx >= N_NODES * HID) return;
  int n = idx >> 7, c = idx & 127;
  float s = 0.f;
#pragma unroll
  for (int f = 0; f < NODE_F; ++f) s += x[n * NODE_F + f] * W1[f * HID + c];
  xw[idx] = s;
}

// ---------------- weight pre-pack into WMMA B-fragment layout --------------
// Fragment layout (16-bit, 32xK per tile): lane = hi*16 + (col&15);
// halves [0..7]  = B[kt*32 + 8*hi + i      ][col]
// halves [8..15] = B[kt*32 + 16 + 8*hi + i ][col]
// packed[((nt*NKT + kt)*32 + lane)*16 + i]

__global__ void k_pack_w2(const float* __restrict__ W2, _Float16* __restrict__ pk) {
  int idx = blockIdx.x * blockDim.x + threadIdx.x;
  if (idx >= NNT * NKT_L2 * 32 * 16) return;
  int i = idx & 15, lane = (idx >> 4) & 31, rest = idx >> 9;
  int kt = rest & 3, nt = rest >> 2;
  int col = nt * 16 + (lane & 15), hi = lane >> 4;
  int k = kt * 32 + (i < 8 ? 8 * hi + i : 16 + 8 * hi + (i - 8));
  pk[idx] = (_Float16)W2[k * HID + col];
}

__global__ void k_pack_wm1(const float* __restrict__ Wm1, _Float16* __restrict__ pk) {
  int idx = blockIdx.x * blockDim.x + threadIdx.x;
  if (idx >= NNT * NKT_MLP * 32 * 16) return;
  int i = idx & 15, lane = (idx >> 4) & 31, rest = idx >> 9;
  int kt = rest % NKT_MLP, nt = rest / NKT_MLP;
  int col = nt * 16 + (lane & 15), hi = lane >> 4;
  int k = kt * 32 + (i < 8 ? 8 * hi + i : 16 + 8 * hi + (i - 8));
  pk[idx] = (k < KCAT) ? (_Float16)Wm1[k * HID + col] : (_Float16)0.f;
}

// ---------------- scatter-aggregate (both GCN layers) ----------------------
// one thread handles (edge, 4 columns); agg[dst] += norm * xw[src]

__global__ void k_scatter(const int* __restrict__ src, const int* __restrict__ dst,
                          const float* __restrict__ nrm, const float* __restrict__ xw,
                          float* __restrict__ agg) {
  int t = blockIdx.x * blockDim.x + threadIdx.x;
  if (t >= N_EDGES * 32) return;
  int e = t >> 5, c = (t & 31) * 4;
  float nm = nrm[e];
  int s = src[e], d = dst[e];
  float4 v = *(const float4*)&xw[(size_t)s * HID + c];
  float* a = &agg[(size_t)d * HID + c];
  unsafeAtomicAdd(a + 0, nm * v.x);
  unsafeAtomicAdd(a + 1, nm * v.y);
  unsafeAtomicAdd(a + 2, nm * v.z);
  unsafeAtomicAdd(a + 3, nm * v.w);
}

// h = [relu](agg + dis^2 * xw + b)  -> f16 node embedding table
__global__ void k_finalize(const float* __restrict__ agg, const float* __restrict__ xw,
                           const float* __restrict__ dis, const float* __restrict__ bias,
                           _Float16* __restrict__ h16, int do_relu) {
  int idx = blockIdx.x * blockDim.x + threadIdx.x;
  if (idx >= N_NODES * HID) return;
  int n = idx >> 7, c = idx & 127;
  float dn = dis[n];
  float v = agg[idx] + dn * dn * xw[idx] + bias[c];
  if (do_relu) v = v > 0.f ? v : 0.f;
  h16[idx] = (_Float16)v;
}

// ---------------- GCN layer-2 GEMM: xw2 = h1 @ W2 (WMMA) -------------------
// one wave per 16-node tile; A from f16 h table, B from packed W2

__global__ void k_gemm_l2(const _Float16* __restrict__ h16,
                          const _Float16* __restrict__ pkW2,
                          float* __restrict__ xw) {
  int wave = (blockIdx.x * blockDim.x + threadIdx.x) >> 5;
  int lane = threadIdx.x & 31;
  if (wave >= N_NODES / 16) return;
  int r15 = lane & 15, hi = lane >> 4;
  int node = wave * 16 + r15;

  v8f acc[NNT] = {};
#pragma unroll
  for (int kt = 0; kt < NKT_L2; ++kt) {
    union { v16h v; v8h h[2]; } a;
    const _Float16* base = h16 + (size_t)node * HID + kt * 32;
    a.h[0] = *(const v8h*)(base + 8 * hi);
    a.h[1] = *(const v8h*)(base + 16 + 8 * hi);
#pragma unroll
    for (int nt = 0; nt < NNT; ++nt) {
      v16h b = *(const v16h*)(pkW2 + ((nt * NKT_L2 + kt) * 32 + lane) * 16);
      acc[nt] = __builtin_amdgcn_wmma_f32_16x16x32_f16(
          false, a.v, false, b, (short)0, acc[nt], false, false);
    }
  }
  // D layout: VGPR r, lanes 0-15 -> M=r ; lanes 16-31 -> M=8+r ; N = lane&15
  int rowBase = wave * 16 + 8 * hi;
#pragma unroll
  for (int nt = 0; nt < NNT; ++nt)
#pragma unroll
    for (int r = 0; r < 8; ++r)
      xw[(size_t)(rowBase + r) * HID + nt * 16 + r15] = acc[nt][r];
}

// ---------------- edge MLP: sigmoid(relu([h_s|h_d|ea] @ Wm1 + bm1) @ Wm2 + bm2)
// one wave per 16-edge tile, 72 WMMAs, fused second layer in epilogue

__global__ void k_edge_mlp(const _Float16* __restrict__ h16,
                           const _Float16* __restrict__ pkWm1,
                           const float* __restrict__ edge_attr,
                           const int* __restrict__ src, const int* __restrict__ dst,
                           const float* __restrict__ bm1, const float* __restrict__ Wm2,
                           const float* __restrict__ bm2, float* __restrict__ out) {
  int wave = (blockIdx.x * blockDim.x + threadIdx.x) >> 5;
  int lane = threadIdx.x & 31;
  if (wave >= N_EDGES / 16) return;
  int r15 = lane & 15, hi = lane >> 4;
  int e = wave * 16 + r15;
  int s = src[e], d = dst[e];

  v8f acc[NNT] = {};
#pragma unroll
  for (int kt = 0; kt < NKT_MLP; ++kt) {
    union { v16h v; v8h h[2]; } a;
    if (kt < 4) {                       // K 0..127  -> h[src]
      const _Float16* base = h16 + (size_t)s * HID + kt * 32;
      a.h[0] = *(const v8h*)(base + 8 * hi);
      a.h[1] = *(const v8h*)(base + 16 + 8 * hi);
    } else if (kt < 8) {                // K 128..255 -> h[dst]
      const _Float16* base = h16 + (size_t)d * HID + (kt - 4) * 32;
      a.h[0] = *(const v8h*)(base + 8 * hi);
      a.h[1] = *(const v8h*)(base + 16 + 8 * hi);
    } else {                            // K 256..287 -> edge_attr + zero pad
#pragma unroll
      for (int j = 0; j < 16; ++j) a.v[j] = (_Float16)0.f;
      if (hi == 0) {
#pragma unroll
        for (int j = 0; j < EDGE_F; ++j)
          a.v[j] = (_Float16)edge_attr[(size_t)e * EDGE_F + j];
      }
    }
#pragma unroll
    for (int nt = 0; nt < NNT; ++nt) {
      v16h b = *(const v16h*)(pkWm1 + ((nt * NKT_MLP + kt) * 32 + lane) * 16);
      acc[nt] = __builtin_amdgcn_wmma_f32_16x16x32_f16(
          false, a.v, false, b, (short)0, acc[nt], false, false);
    }
  }

  // fused layer 2: per-lane partials over its 8 columns, then 16-lane reduce
  float bmv[NNT], wmv[NNT];
#pragma unroll
  for (int nt = 0; nt < NNT; ++nt) {
    int col = nt * 16 + r15;
    bmv[nt] = bm1[col];
    wmv[nt] = Wm2[col];
  }
  float b2s = bm2[0];
#pragma unroll
  for (int r = 0; r < 8; ++r) {
    float p = 0.f;
#pragma unroll
    for (int nt = 0; nt < NNT; ++nt) {
      float v = acc[nt][r] + bmv[nt];
      v = v > 0.f ? v : 0.f;
      p += v * wmv[nt];
    }
    p += __shfl_xor(p, 1);
    p += __shfl_xor(p, 2);
    p += __shfl_xor(p, 4);
    p += __shfl_xor(p, 8);   // stays within each 16-lane half (wave32)
    if (r15 == 0) {
      int eo = wave * 16 + 8 * hi + r;
      out[eo] = 1.f / (1.f + expf(-(p + b2s)));
    }
  }
}

// ---------------------------------------------------------------------------

extern "C" void kernel_launch(void* const* d_in, const int* in_sizes, int n_in,
                              void* d_out, int out_size, void* d_ws, size_t ws_size,
                              hipStream_t stream) {
  (void)in_sizes; (void)n_in; (void)out_size; (void)ws_size;

  const float* x         = (const float*)d_in[0];
  const float* edge_attr = (const float*)d_in[1];
  const float* W1        = (const float*)d_in[2];
  const float* b1        = (const float*)d_in[3];
  const float* W2        = (const float*)d_in[4];
  const float* b2        = (const float*)d_in[5];
  const float* Wm1       = (const float*)d_in[6];
  const float* bm1       = (const float*)d_in[7];
  const float* Wm2       = (const float*)d_in[8];
  const float* bm2       = (const float*)d_in[9];
  const int*   eidx      = (const int*)d_in[10];
  const int*   src       = eidx;            // edge_index[0]
  const int*   dst       = eidx + N_EDGES;  // edge_index[1]
  float* out = (float*)d_out;

  // workspace layout (all segments 32B-aligned)
  char* ws = (char*)d_ws;
  float*    dis   = (float*)(ws);                        // N      (deg -> dis)
  float*    nrm   = (float*)(ws + 200000);               // E
  float*    xw    = (float*)(ws + 3400000);              // N*H
  float*    agg   = (float*)(ws + 29000000);             // N*H
  _Float16* h16   = (_Float16*)(ws + 54600000);          // N*H (f16)
  _Float16* pkW2  = (_Float16*)(ws + 67400000);          // 8*4*32*16
  _Float16* pkWm1 = (_Float16*)(ws + 67432768);          // 8*9*32*16

  const int NH = N_NODES * HID;
  const int T  = 256;
  auto blks = [](int n, int t) { return (n + t - 1) / t; };

  // degree / normalization
  k_zero<<<blks(N_NODES, T), T, 0, stream>>>(dis, N_NODES);
  k_zero<<<blks(NH, T), T, 0, stream>>>(agg, NH);
  k_deg<<<blks(N_EDGES, T), T, 0, stream>>>(dst, dis);
  k_dis<<<blks(N_NODES, T), T, 0, stream>>>(dis);
  k_norm<<<blks(N_EDGES, T), T, 0, stream>>>(src, dst, dis, nrm);

  // weight pre-pack (WMMA B-fragment layout)
  k_pack_w2<<<blks(NNT * NKT_L2 * 32 * 16, T), T, 0, stream>>>(W2, pkW2);
  k_pack_wm1<<<blks(NNT * NKT_MLP * 32 * 16, T), T, 0, stream>>>(Wm1, pkWm1);

  // GCN layer 1
  k_xw1<<<blks(NH, T), T, 0, stream>>>(x, W1, xw);
  k_scatter<<<blks(N_EDGES * 32, T), T, 0, stream>>>(src, dst, nrm, xw, agg);
  k_finalize<<<blks(NH, T), T, 0, stream>>>(agg, xw, dis, b1, h16, /*relu=*/1);

  // GCN layer 2 (WMMA GEMM + scatter)
  k_zero<<<blks(NH, T), T, 0, stream>>>(agg, NH);
  k_gemm_l2<<<blks((N_NODES / 16) * 32, T), T, 0, stream>>>(h16, pkW2, xw);
  k_scatter<<<blks(N_EDGES * 32, T), T, 0, stream>>>(src, dst, nrm, xw, agg);
  k_finalize<<<blks(NH, T), T, 0, stream>>>(agg, xw, dis, b2, h16, /*relu=*/0);

  // fused edge MLP (WMMA) + sigmoid
  k_edge_mlp<<<blks((N_EDGES / 16) * 32, T), T, 0, stream>>>(
      h16, pkWm1, edge_attr, src, dst, bm1, Wm2, bm2, out);
}
